// QNetGNN_68977174774273
// MI455X (gfx1250) — matrix operand order repaired and checked by reading, not compiled, not run on hardware
//
#include <hip/hip_runtime.h>

typedef __attribute__((ext_vector_type(16))) _Float16 v16h;
typedef __attribute__((ext_vector_type(8)))  float    v8f;

#define F_IN  128
#define F_HID 16
#define F_ACT 32

// ---------------------------------------------------------------- utilities
__global__ void init_deg_kernel(float* __restrict__ deg, int n) {
  int i = blockIdx.x * blockDim.x + threadIdx.x;
  if (i < n) deg[i] = 1.0f;                    // self-loop contributes 1
}

__global__ void zero_kernel(float* __restrict__ p, long long n) {
  long long i = (long long)blockIdx.x * blockDim.x + threadIdx.x;
  if (i < n) p[i] = 0.0f;
}

__global__ void count_deg_kernel(const long long* __restrict__ ei,
                                 float* __restrict__ deg, long long nE) {
  long long e = (long long)blockIdx.x * blockDim.x + threadIdx.x;
  if (e < nE) {
    int d = (int)ei[nE + e];                   // dst row of edge_index
    atomicAdd(&deg[d], 1.0f);
  }
}

__global__ void rsqrt_kernel(const float* __restrict__ deg,
                             float* __restrict__ dis, int n) {
  int i = blockIdx.x * blockDim.x + threadIdx.x;
  if (i < n) dis[i] = rsqrtf(deg[i]);          // deg >= 1 always (self-loops)
}

// ------------------------------------------------------- GEMM1: x @ W1 (K=128)
// One 16-row tile per wave. A: 16x32 f16 fragment per K-step, B: 32x16 f16.
__global__ void gemm1_wmma(const float* __restrict__ x,
                           const float* __restrict__ W1,
                           float* __restrict__ h1, int n) {
  const int lane = threadIdx.x & 31;
  const int wave = threadIdx.x >> 5;
  const int tile = blockIdx.x * 2 + wave;      // 2 waves / block
  const int row0 = tile << 4;
  if (row0 >= n) return;                       // wave-uniform exit (EXEC stays full)
  const int m    = lane & 15;
  const int half = lane >> 4;                  // lane group 0-15 / 16-31
  const float* arow = x + (long long)(row0 + m) * F_IN;

  v8f c = {0.f, 0.f, 0.f, 0.f, 0.f, 0.f, 0.f, 0.f};
#pragma unroll
  for (int kk = 0; kk < F_IN; kk += 32) {
    v16h a, b;
#pragma unroll
    for (int v = 0; v < 8; ++v) {
      // A (16-bit 16x32): VGPR v holds K = (v>>2)*16 + half*8 + (v&3)*2 .. +1
      int ka = kk + ((v >> 2) << 4) + (half << 3) + ((v & 3) << 1);
      a[2 * v]     = (_Float16)arow[ka];
      a[2 * v + 1] = (_Float16)arow[ka + 1];
      // B (32x16): same K order as A: VGPR v holds K = (v>>2)*16 + half*8 + (v&3)*2 .. +1
      int kb = kk + ((v >> 2) << 4) + (half << 3) + ((v & 3) << 1);
      b[2 * v]     = (_Float16)W1[kb * F_HID + m];
      b[2 * v + 1] = (_Float16)W1[(kb + 1) * F_HID + m];
    }
    c = __builtin_amdgcn_wmma_f32_16x16x32_f16(false, a, false, b,
                                               (short)0, c, false, false);
  }
  // C/D: VGPR v -> row (v + half*8), col = lane&15
#pragma unroll
  for (int v = 0; v < 8; ++v)
    h1[(long long)(row0 + (half << 3) + v) * F_HID + m] = c[v];
}

// ------------------------------------------- GEMM2: h1r @ W2 (K=16 pad to 32)
__global__ void gemm2_wmma(const float* __restrict__ h1r,
                           const float* __restrict__ W2,
                           float* __restrict__ h2, int n) {
  const int lane = threadIdx.x & 31;
  const int wave = threadIdx.x >> 5;
  const int tile = blockIdx.x * 2 + wave;
  const int row0 = tile << 4;
  if (row0 >= n) return;
  const int m    = lane & 15;
  const int half = lane >> 4;
  const float* arow = h1r + (long long)(row0 + m) * F_HID;

  v16h a;
#pragma unroll
  for (int v = 0; v < 8; ++v) {
    int ka = ((v >> 2) << 4) + (half << 3) + ((v & 3) << 1);   // 0..31
    bool in = ka < F_HID;                                      // pad K>=16 with 0
    a[2 * v]     = in ? (_Float16)arow[ka]     : (_Float16)0.f;
    a[2 * v + 1] = in ? (_Float16)arow[ka + 1] : (_Float16)0.f;
  }
#pragma unroll
  for (int nt = 0; nt < 2; ++nt) {             // two 16-col output tiles (N=32)
    v16h b;
#pragma unroll
    for (int v = 0; v < 8; ++v) {
      int kb = ((v >> 2) << 4) + (half << 3) + ((v & 3) << 1);
      bool in = kb < F_HID;                    // K>=16 (v>=4) -> 0
      b[2 * v]     = in ? (_Float16)W2[kb * F_ACT + nt * 16 + m]       : (_Float16)0.f;
      b[2 * v + 1] = in ? (_Float16)W2[(kb + 1) * F_ACT + nt * 16 + m] : (_Float16)0.f;
    }
    v8f c = {0.f, 0.f, 0.f, 0.f, 0.f, 0.f, 0.f, 0.f};
    c = __builtin_amdgcn_wmma_f32_16x16x32_f16(false, a, false, b,
                                               (short)0, c, false, false);
#pragma unroll
    for (int v = 0; v < 8; ++v)
      h2[(long long)(row0 + (half << 3) + v) * F_ACT + nt * 16 + m] = c[v];
  }
}

// --------------------------------------------------- edge scatter (atomic add)
template <int F, int LOGF>
__global__ void scatter_kernel(const long long* __restrict__ ei,
                               const float* __restrict__ dis,
                               const float* __restrict__ h,
                               float* __restrict__ agg, long long nE) {
  long long t = (long long)blockIdx.x * blockDim.x + threadIdx.x;
  if (t >= (nE << LOGF)) return;
  long long e = t >> LOGF;
  int f = (int)(t & (F - 1));
  int s = (int)ei[e];
  int d = (int)ei[nE + e];
  float nm = dis[s] * dis[d];
  atomicAdd(&agg[(long long)d * F + f], h[(long long)s * F + f] * nm);
}

// ----------------------- finalize: self-loop term + bias (+ optional ReLU)
template <int F, bool RELU>
__global__ void finalize_kernel(const float* __restrict__ agg,
                                const float* __restrict__ h,
                                const float* __restrict__ dis,
                                const float* __restrict__ bias,
                                float* __restrict__ outp, int n) {
  long long t = (long long)blockIdx.x * blockDim.x + threadIdx.x;
  if (t >= (long long)n * F) return;
  int i = (int)(t >> (F == 16 ? 4 : 5));
  int f = (int)(t & (F - 1));
  float di = dis[i];
  float v = agg[t] + h[t] * di * di + bias[f];
  outp[t] = RELU ? fmaxf(v, 0.0f) : v;
}

// ---------------------------------------------------------------- launcher
extern "C" void kernel_launch(void* const* d_in, const int* in_sizes, int n_in,
                              void* d_out, int out_size, void* d_ws, size_t ws_size,
                              hipStream_t stream) {
  (void)n_in; (void)out_size; (void)ws_size;

  const float*     x  = (const float*)d_in[0];
  const float*     W1 = (const float*)d_in[1];
  const float*     b1 = (const float*)d_in[2];
  const float*     W2 = (const float*)d_in[3];
  const float*     b2 = (const float*)d_in[4];
  const long long* ei = (const long long*)d_in[5];   // int64 [2, E]

  const int       n  = in_sizes[0] / F_IN;           // 100000
  const long long nE = (long long)in_sizes[5] / 2;   // 3200000
  float* outp = (float*)d_out;

  // workspace carve-up (fp32)
  float* deg  = (float*)d_ws;                        // [n]
  float* dis  = deg  + n;                            // [n]
  float* h1   = dis  + n;                            // [n,16]
  float* agg1 = h1   + (long long)n * F_HID;         // [n,16]
  float* h1r  = agg1 + (long long)n * F_HID;         // [n,16]
  float* h2   = h1r  + (long long)n * F_HID;         // [n,32]
  float* agg2 = h2   + (long long)n * F_ACT;         // [n,32]

  const int B = 256;
  dim3 blk(B);

  // 1) degrees (self-loop = 1) + zeroed accumulators + norms
  init_deg_kernel<<<dim3((n + B - 1) / B), blk, 0, stream>>>(deg, n);
  zero_kernel<<<dim3((unsigned)(((long long)n * F_HID + B - 1) / B)), blk, 0, stream>>>(agg1, (long long)n * F_HID);
  zero_kernel<<<dim3((unsigned)(((long long)n * F_ACT + B - 1) / B)), blk, 0, stream>>>(agg2, (long long)n * F_ACT);
  count_deg_kernel<<<dim3((unsigned)((nE + B - 1) / B)), blk, 0, stream>>>(ei, deg, nE);
  rsqrt_kernel<<<dim3((n + B - 1) / B), blk, 0, stream>>>(deg, dis, n);

  // 2) layer 1 projection (WMMA), tiles of 16 rows, 2 waves/block
  const int tiles = (n + 15) / 16;
  gemm1_wmma<<<dim3((tiles + 1) / 2), dim3(64), 0, stream>>>(x, W1, h1, n);

  // 3) layer 1 aggregation + finalize (bias, self-loop, ReLU)
  scatter_kernel<F_HID, 4><<<dim3((unsigned)(((nE << 4) + B - 1) / B)), blk, 0, stream>>>(ei, dis, h1, agg1, nE);
  finalize_kernel<F_HID, true><<<dim3((unsigned)(((long long)n * F_HID + B - 1) / B)), blk, 0, stream>>>(agg1, h1, dis, b1, h1r, n);

  // 4) layer 2 projection (WMMA, K=16 zero-padded to 32, N=32 as two tiles)
  gemm2_wmma<<<dim3((tiles + 1) / 2), dim3(64), 0, stream>>>(h1r, W2, h2, n);

  // 5) layer 2 aggregation + finalize (bias, self-loop) -> d_out
  scatter_kernel<F_ACT, 5><<<dim3((unsigned)(((nE << 5) + B - 1) / B)), blk, 0, stream>>>(ei, dis, h2, agg2, nE);
  finalize_kernel<F_ACT, false><<<dim3((unsigned)(((long long)n * F_ACT + B - 1) / B)), blk, 0, stream>>>(agg2, h2, dis, b2, outp, n);
}
